// Ex4bitLinear_65738769432997
// MI455X (gfx1250) — compile-verified, matrix operand order checked by reading, not run on hardware
//
#include <hip/hip_runtime.h>

// GPTQ 4-bit dequant + GEMM for gfx1250 (MI455X), bf16 WMMA path.
//   out[T,N] = x[T,K] @ dequant(qweight,qzeros,scales)[K,N] + bias
// Compute-bound: maximize v_wmma density. Block tile 128x256, 8 waves (wave32)
// in a 2x4 grid, each wave owns a 64x64 patch = 16 accumulators (128 VGPRs).
#define K_DIM   4096
#define N_DIM   4096
#define GROUPSZ 128
#define BM      128
#define BN      256
#define BK      32      // one WMMA K-step per stage
#define KPAD    48      // pitch in ushorts -> 96B = 3*32B, keeps fragments 32B aligned
#define THREADS 256     // 8 waves

typedef __attribute__((ext_vector_type(16))) __bf16 v16bf;
typedef __attribute__((ext_vector_type(2)))  __bf16 v2bf;
typedef __attribute__((ext_vector_type(8)))  float  v8f;

__device__ __forceinline__ unsigned int pack_bf16(float a, float b) {
  v2bf v;
  v.x = (__bf16)a;           // RNE convert; backend can fuse to v_cvt_pk_bf16_f32
  v.y = (__bf16)b;
  union { v2bf v; unsigned int u; } c;
  c.v = v;
  return c.u;
}

__global__ __launch_bounds__(THREADS, 1)
void gptq_bf16_wmma_kernel(const float* __restrict__ x,
                           const int*   __restrict__ qweight,
                           const int*   __restrict__ qzeros,
                           const float* __restrict__ scales,
                           const float* __restrict__ bias,
                           float*       __restrict__ out,
                           int tokens) {
  __shared__ unsigned short As[BM][KPAD];  // x tile, bf16, [m][k]
  __shared__ unsigned short Bs[BN][KPAD];  // dequant W tile, bf16, [n][k] (n-major)

  const int tid  = threadIdx.x;
  const int lane = tid & 31;
  const int wave = tid >> 5;

  const int block_n = blockIdx.x * BN;
  const int block_m = blockIdx.y * BM;
  if (block_m >= tokens) return;

  // A-load mapping: 2 threads per M-row, each covers 16 contiguous K (f32)
  const int a_row  = tid >> 1;          // 0..127
  const int a_koff = (tid & 1) * 16;    // 0 or 16

  // B-dequant mapping: 4 packed-K rows x 64 col-groups of 4 N
  const int b_kp = tid >> 6;            // packed row within tile: 0..3 (k = b_kp*8..+7)
  const int b_n0 = (tid & 63) * 4;      // N offset within tile

  // wave tiling: 2 (M) x 4 (N) waves, each computes 64x64
  const int wm = (wave & 1) * 64;
  const int wn = (wave >> 1) * 64;

  v8f acc[4][4];
#pragma unroll
  for (int i = 0; i < 4; ++i)
#pragma unroll
    for (int j = 0; j < 4; ++j)
      acc[i][j] = (v8f){0.f, 0.f, 0.f, 0.f, 0.f, 0.f, 0.f, 0.f};

  const float* xrow = x + (size_t)(block_m + a_row) * K_DIM + a_koff;

  for (int k0 = 0; k0 < K_DIM; k0 += BK) {
    // ---------------- global -> registers ----------------
    float4 af[4];
#pragma unroll
    for (int i = 0; i < 4; ++i)
      af[i] = *reinterpret_cast<const float4*>(xrow + k0 + i * 4);

    const int g   = k0 / GROUPSZ;       // whole BK tile lies in one group
    const int kp0 = k0 >> 3;
    const int4 qw = *reinterpret_cast<const int4*>(
        qweight + (size_t)(kp0 + b_kp) * N_DIM + block_n + b_n0);
    const float4 sc = *reinterpret_cast<const float4*>(
        scales + (size_t)g * N_DIM + block_n + b_n0);
    const int zword = qzeros[(size_t)g * (N_DIM / 8) + (block_n + b_n0) / 8];
    const int zsh   = (block_n + b_n0) & 7;   // 0 or 4

    if (k0 + BK < K_DIM) {               // CDNA5 global_prefetch_b8
      __builtin_prefetch(xrow + k0 + BK, 0, 3);
      __builtin_prefetch(qweight + (size_t)((k0 + BK) >> 3) * N_DIM + block_n + b_n0, 0, 3);
    }

    __syncthreads();   // previous tile fully consumed

    // ---------------- A: f32 -> bf16 into LDS ----------------
#pragma unroll
    for (int i = 0; i < 2; ++i) {
      unsigned int p0 = pack_bf16(af[2*i].x,   af[2*i].y);
      unsigned int p1 = pack_bf16(af[2*i].z,   af[2*i].w);
      unsigned int p2 = pack_bf16(af[2*i+1].x, af[2*i+1].y);
      unsigned int p3 = pack_bf16(af[2*i+1].z, af[2*i+1].w);
      *reinterpret_cast<uint4*>(&As[a_row][a_koff + i * 8]) = make_uint4(p0, p1, p2, p3);
    }

    // ---------------- B: 4-bit dequant -> bf16 into LDS [n][k] ----------------
    {
      const int   qv[4] = {qw.x, qw.y, qw.z, qw.w};
      const float sv[4] = {sc.x, sc.y, sc.z, sc.w};
#pragma unroll
      for (int j = 0; j < 4; ++j) {
        const float zb = (float)(((zword >> ((zsh + j) * 4)) & 0xF) + 1);
        unsigned int pk[4];
#pragma unroll
        for (int p = 0; p < 4; ++p) {
          float w0 = (float)((qv[j] >> (8 * p))     & 0xF) - zb;
          float w1 = (float)((qv[j] >> (8 * p + 4)) & 0xF) - zb;
          pk[p] = pack_bf16(sv[j] * w0, sv[j] * w1);
        }
        *reinterpret_cast<uint4*>(&Bs[b_n0 + j][b_kp * 8]) =
            make_uint4(pk[0], pk[1], pk[2], pk[3]);
      }
    }

    __syncthreads();   // tile visible to all waves

    // ---------------- WMMA compute: 16 x v_wmma per wave per stage ----------------
    {
      const int khalf = (lane >> 4) << 4;   // lanes 0-15: K lo half, 16-31: hi half
      v16bf afr[4], bfr[4];
#pragma unroll
      for (int mi = 0; mi < 4; ++mi)
        afr[mi] = *reinterpret_cast<const v16bf*>(&As[wm + mi * 16 + (lane & 15)][khalf]);
#pragma unroll
      for (int ni = 0; ni < 4; ++ni)
        bfr[ni] = *reinterpret_cast<const v16bf*>(&Bs[wn + ni * 16 + (lane & 15)][khalf]);
#pragma unroll
      for (int mi = 0; mi < 4; ++mi)
#pragma unroll
        for (int ni = 0; ni < 4; ++ni)
          acc[mi][ni] = __builtin_amdgcn_wmma_f32_16x16x32_bf16(
              false, afr[mi], false, bfr[ni], (short)0, acc[mi][ni], false, false);
    }
  }

  // ---------------- epilogue: C layout (VGPR r -> M = r + 8*(lane>=16), N = lane%16) ---
#pragma unroll
  for (int ni = 0; ni < 4; ++ni) {
    const int col = block_n + wn + ni * 16 + (lane & 15);
    const float bv = bias[col];
#pragma unroll
    for (int mi = 0; mi < 4; ++mi) {
      const int rbase = block_m + wm + mi * 16 + ((lane >> 4) << 3);
#pragma unroll
      for (int r = 0; r < 8; ++r)
        out[(size_t)(rbase + r) * N_DIM + col] = acc[mi][ni][r] + bv;
    }
  }
}

extern "C" void kernel_launch(void* const* d_in, const int* in_sizes, int n_in,
                              void* d_out, int out_size, void* d_ws, size_t ws_size,
                              hipStream_t stream) {
  (void)n_in; (void)out_size; (void)d_ws; (void)ws_size;
  const float* x       = (const float*)d_in[0];
  const int*   qweight = (const int*)  d_in[1];
  const int*   qzeros  = (const int*)  d_in[2];
  const float* scales  = (const float*)d_in[3];
  // d_in[4] = g_idx: standard (non act-order) grouping, folded into k/GROUPSZ.
  const float* bias    = (const float*)d_in[5];
  float*       out     = (float*)d_out;

  const int tokens = in_sizes[0] / K_DIM;   // 8192
  dim3 grid(N_DIM / BN, tokens / BM);       // (16, 64)
  gptq_bf16_wmma_kernel<<<grid, THREADS, 0, stream>>>(
      x, qweight, qzeros, scales, bias, out, tokens);
}
